// DecoderLayer_27358941675735
// MI455X (gfx1250) — compile-verified, matrix operand order checked by reading, not run on hardware
//
#include <hip/hip_runtime.h>
#include <hip/hip_bf16.h>
#include <hip/hip_fp16.h>

// Problem constants (match reference)
#define B_   2
#define Q_   300
#define K_   1024
#define D_   256
#define H_   8
#define HD_  32
#define E_   131072
#define N_   600      // B*Q
#define CPB_ 256

typedef __attribute__((ext_vector_type(16))) _Float16 v16h;
typedef __attribute__((ext_vector_type(8)))  _Float16 v8h_t;
typedef __attribute__((ext_vector_type(8)))  float    v8f;
typedef __attribute__((ext_vector_type(4)))  unsigned int v4u;
typedef __attribute__((ext_vector_type(8)))  int v8i;
typedef __attribute__((ext_vector_type(4)))  int v4i;

// ---------------------------------------------------------------------------
// WMMA tile loaders (CDNA5 wave32 16x16x32 f16 layout, ISA 7.12.2):
//   A (16x32, M x K) : lane = row(l&15); lanes<16 hold K {0..7,16..23},
//                      lanes>=16 hold K {8..15,24..31}  -> two 16B chunks.
//   B (32x16, K x N) : identical pattern with N-major storage (i.e. B^T
//                      row-major), so the same loader works for both.
// Caller passes p = &src[row*ld + k0 + kb] with kb = (lane>>4)*8.
// ---------------------------------------------------------------------------
__device__ __forceinline__ v16h ld_tile_f32(const float* __restrict__ p) {
  float4 f0 = *(const float4*)(p);
  float4 f1 = *(const float4*)(p + 4);
  float4 f2 = *(const float4*)(p + 16);
  float4 f3 = *(const float4*)(p + 20);
  v16h a;
  a[0]=(_Float16)f0.x;  a[1]=(_Float16)f0.y;  a[2]=(_Float16)f0.z;  a[3]=(_Float16)f0.w;
  a[4]=(_Float16)f1.x;  a[5]=(_Float16)f1.y;  a[6]=(_Float16)f1.z;  a[7]=(_Float16)f1.w;
  a[8]=(_Float16)f2.x;  a[9]=(_Float16)f2.y;  a[10]=(_Float16)f2.z; a[11]=(_Float16)f2.w;
  a[12]=(_Float16)f3.x; a[13]=(_Float16)f3.y; a[14]=(_Float16)f3.z; a[15]=(_Float16)f3.w;
  return a;
}

__device__ __forceinline__ v16h ld_tile_f16(const _Float16* p) {
  v8h_t x0 = *(const v8h_t*)(p);
  v8h_t x1 = *(const v8h_t*)(p + 16);
  v16h a;
#pragma unroll
  for (int i = 0; i < 8; i++) { a[i] = x0[i]; a[8+i] = x1[i]; }
  return a;
}

__device__ __forceinline__ v8f wmma_f16(v16h a, v16h b, v8f c) {
  return __builtin_amdgcn_wmma_f32_16x16x32_f16(false, a, false, b, (short)0, c,
                                                false, false);
}

// ---------------------------------------------------------------------------
// Tensor Data Mover: DMA a 2D f16 tile (width x rows, row stride in elems)
// from global into LDS (compacted, ld = width).  D# layout per ISA ch.8.
// One instruction moves the whole panel; completion via TENSORcnt.
// This toolchain exposes the 6-arg builtin:
//   (uint32x4 g0, int32x8 g1, int32x4 g2, int32x4 g3, int32x8 pad, i32 cpol)
// ---------------------------------------------------------------------------
#if __has_builtin(__builtin_amdgcn_tensor_load_to_lds)
#define TDM_AVAILABLE 1
__device__ __forceinline__ void tdm_load_f16_tile(unsigned lds_off,
                                                  const _Float16* gptr,
                                                  unsigned width, unsigned rows,
                                                  unsigned stride_elems) {
  unsigned long long ga = (unsigned long long)(uintptr_t)gptr;
  v4u g0;
  g0[0] = 1u;                                   // count=1, user descriptor
  g0[1] = lds_off;                              // lds_addr (bytes)
  g0[2] = (unsigned)(ga & 0xFFFFFFFFull);       // global_addr[31:0]
  g0[3] = (unsigned)((ga >> 32) & 0x01FFFFFFull) | (2u << 30); // [56:32]|type=2
  v8i g1;
  g1[0] = (int)(1u << 16);                      // data_size=1 (2 bytes)
  g1[1] = (int)((width & 0xFFFFu) << 16);       // tensor_dim0[15:0]
  g1[2] = (int)(((width >> 16) & 0xFFFFu) | ((rows & 0xFFFFu) << 16)); // td0 hi | td1 lo
  g1[3] = (int)(((rows >> 16) & 0xFFFFu) | ((width & 0xFFFFu) << 16)); // td1 hi | tile_dim0
  g1[4] = (int)(rows & 0xFFFFu);                // tile_dim1 (tile_dim2 = 0)
  g1[5] = (int)stride_elems;                    // tensor_dim0_stride low32
  g1[6] = 0;                                    // stride0 hi | dim1_stride lo
  g1[7] = 0;
  v4i z4 = {0, 0, 0, 0};
  v8i z8 = {0, 0, 0, 0, 0, 0, 0, 0};
  __builtin_amdgcn_tensor_load_to_lds(g0, g1, z4, z4, z8, 0);
}
#else
#define TDM_AVAILABLE 0
#endif

// Fallback cooperative stage (also keeps kernels compiling w/o TDM builtin)
__device__ __forceinline__ void lds_stage_f16(_Float16* sW, const _Float16* g,
                                              int width, int rows, int stride) {
  int chunks = rows * (width / 8);
  for (int i = threadIdx.x; i < chunks; i += 256) {
    int r = i / (width / 8);
    int c = (i - r * (width / 8)) * 8;
    *(v8h_t*)(sW + r * width + c) = *(const v8h_t*)(g + (size_t)r * stride + c);
  }
}

// Stage a (width=128) x (rows=256) f16 panel into sW and sync the block.
__device__ __forceinline__ void stage_panel(_Float16* sW, const _Float16* g,
                                            unsigned stride_elems) {
#if TDM_AVAILABLE
  if (threadIdx.x == 0) {
    tdm_load_f16_tile((unsigned)(uintptr_t)(void*)sW, g, 128u, 256u, stride_elems);
  }
  __builtin_amdgcn_s_wait_tensorcnt(0);
  __syncthreads();
#else
  lds_stage_f16(sW, g, 128, 256, (int)stride_elems);
  __syncthreads();
#endif
}

// ---------------------------------------------------------------------------
// Weight prep: f32 (K x N) row-major  ->  f16 (N x K) row-major ("Bt")
// ---------------------------------------------------------------------------
__global__ __launch_bounds__(256) void k_transpose_cvt(
    const float* __restrict__ src, _Float16* __restrict__ dst, int K, int N) {
  int idx = blockIdx.x * 256 + threadIdx.x;
  if (idx >= K * N) return;
  int k = idx / N, n = idx - k * N;
  dst[(size_t)n * K + k] = (_Float16)src[idx];
}

__global__ void k_add3(const float* a, const float* b, const float* c, float* o) {
  int i = threadIdx.x;
  o[i] = a[i] + b[i] + c[i];
}

// ---------------------------------------------------------------------------
// LayerNorm over D=256, one block per row.
// ---------------------------------------------------------------------------
__global__ __launch_bounds__(256) void k_ln(const float* __restrict__ x,
                                            const float* __restrict__ g,
                                            const float* __restrict__ bb,
                                            float* __restrict__ y) {
  int row = blockIdx.x, t = threadIdx.x;
  float v = x[(size_t)row * 256 + t];
  float s = v, q = v * v;
#pragma unroll
  for (int m = 1; m < 32; m <<= 1) {
    s += __shfl_xor(s, m, 32);
    q += __shfl_xor(q, m, 32);
  }
  __shared__ float sm[8], sq[8];
  int wid = t >> 5, ln = t & 31;
  if (ln == 0) { sm[wid] = s; sq[wid] = q; }
  __syncthreads();
  float ts = 0.f, tq = 0.f;
#pragma unroll
  for (int i = 0; i < 8; i++) { ts += sm[i]; tq += sq[i]; }
  float mu  = ts * (1.0f / 256.0f);
  float var = tq * (1.0f / 256.0f) - mu * mu;
  float inv = rsqrtf(var + 1e-5f);
  y[(size_t)row * 256 + t] = (v - mu) * inv * g[t] + bb[t];
}

// ---------------------------------------------------------------------------
// Generic WMMA GEMM: C[M x Ncols] = A[M x K]f32 @ B + bias (+resid) (+gelu)
// Bt is f16 N-major (N x K).  One wave computes a 16-row x 256-col panel.
// ---------------------------------------------------------------------------
__global__ __launch_bounds__(256) void k_gemm(
    const float* __restrict__ A, int lda,
    const _Float16* __restrict__ Bt, int ldb,
    const float* __restrict__ bias,
    const float* __restrict__ resid, int ldr,
    float* __restrict__ C, int ldc,
    int M, int npanels, int ksteps, int act) {
  int wid = threadIdx.x >> 5, lane = threadIdx.x & 31;
  int gw = blockIdx.x * 8 + wid;
  int mt = gw / npanels, np = gw - mt * npanels;
  int m0 = mt * 16;
  if (m0 >= M) return;
  int ln = lane & 15, hi = lane >> 4, kb = hi * 8;
  int rowA = m0 + ln; if (rowA > M - 1) rowA = M - 1;
  v8f acc[16] = {};
  for (int ks = 0; ks < ksteps; ks++) {
    int k0 = ks * 32;
    v16h a = ld_tile_f32(A + (size_t)rowA * lda + k0 + kb);
#pragma unroll
    for (int n = 0; n < 16; n++) {
      int rb = np * 256 + n * 16 + ln;
      v16h b = ld_tile_f16(Bt + (size_t)rb * ldb + k0 + kb);
      acc[n] = wmma_f16(a, b, acc[n]);
    }
  }
#pragma unroll
  for (int n = 0; n < 16; n++) {
    int col = np * 256 + n * 16 + ln;
    float bv = bias ? bias[col] : 0.0f;
#pragma unroll
    for (int r = 0; r < 8; r++) {
      int m = m0 + r + 8 * hi;
      if (m >= M) continue;
      float v = acc[n][r] + bv;
      if (resid) v += resid[(size_t)m * ldr + col];
      if (act == 1) v = 0.5f * v * (1.0f + erff(v * 0.70710678118654752f));
      C[(size_t)m * ldc + col] = v;
    }
  }
}

// ---------------------------------------------------------------------------
// Edge pass 1: hidden = leaky( ef@We + Pi[src] + Pj[dst] )  (bias folded in Pi)
// then logits[e,h] = <hidden[e, 32h:32h+32], attn_proj[h]>
// One wave = 16 consecutive edges x 256 cols; We staged in LDS via TDM in two
// 64 KB half-panels (K 0..127, 128..255) shared by all 8 waves of the block.
// ---------------------------------------------------------------------------
__global__ __launch_bounds__(256) void k_edge_hidden(
    const float* __restrict__ ef, const _Float16* __restrict__ WeT,
    const float* __restrict__ Pi, const float* __restrict__ Pj,
    const int* __restrict__ ei, const float* __restrict__ attn_proj,
    float* __restrict__ logits) {
  __shared__ _Float16 sW[256 * 128];  // 64 KB
  int wid = threadIdx.x >> 5, lane = threadIdx.x & 31;
  int e0 = (blockIdx.x * 8 + wid) * 16;
  int ln = lane & 15, hi = lane >> 4, kb = hi * 8;
  const float* arow = ef + (size_t)(e0 + ln) * 256;
  v8f acc[16] = {};
  for (int phase = 0; phase < 2; ++phase) {
    if (phase) __syncthreads();  // previous-phase readers done before overwrite
    stage_panel(sW, WeT + phase * 128, 256u);
    for (int ks = 0; ks < 4; ks++) {
      int k0 = phase * 128 + ks * 32;
      __builtin_prefetch(arow + 128 * 256 + k0, 0, 3);  // next block's rows
      v16h a = ld_tile_f32(arow + k0 + kb);
#pragma unroll
      for (int n = 0; n < 16; n++) {
        v16h b = ld_tile_f16(sW + (size_t)(n * 16 + ln) * 128 + ks * 32 + kb);
        acc[n] = wmma_f16(a, b, acc[n]);
      }
    }
  }
#pragma unroll
  for (int r = 0; r < 8; r++) {
    int e = e0 + r + 8 * hi;
    int s = ei[e], d = ei[E_ + e];
    float hv[16];
#pragma unroll
    for (int n = 0; n < 16; n++) {
      int col = n * 16 + ln;
      float x = acc[n][r] + Pi[(size_t)s * 256 + col] + Pj[(size_t)d * 256 + col];
      hv[n] = (x > 0.0f) ? x : 0.2f * x;
    }
#pragma unroll
    for (int h = 0; h < 8; h++) {
      float part = hv[2 * h]     * attn_proj[h * 32 + ln]
                 + hv[2 * h + 1] * attn_proj[h * 32 + 16 + ln];
      part += __shfl_xor(part, 1, 32);
      part += __shfl_xor(part, 2, 32);
      part += __shfl_xor(part, 4, 32);
      part += __shfl_xor(part, 8, 32);
      if (ln == 0) logits[(size_t)e * 8 + h] = part;
    }
  }
}

// ---------------------------------------------------------------------------
// Edge pass 2: msg = ef@Wmsg[256:512] + Mj[dst]   (bmsg folded in Mj), f16 out
// Strided sub-panel of Wmsg^T staged via TDM (row stride 512 elems).
// ---------------------------------------------------------------------------
__global__ __launch_bounds__(256) void k_edge_msg(
    const float* __restrict__ ef, const _Float16* __restrict__ WmT,
    const float* __restrict__ Mj, const int* __restrict__ ei,
    _Float16* __restrict__ msg) {
  __shared__ _Float16 sW[256 * 128];  // 64 KB
  int wid = threadIdx.x >> 5, lane = threadIdx.x & 31;
  int e0 = (blockIdx.x * 8 + wid) * 16;
  int ln = lane & 15, hi = lane >> 4, kb = hi * 8;
  const float* arow = ef + (size_t)(e0 + ln) * 256;
  v8f acc[16] = {};
  for (int phase = 0; phase < 2; ++phase) {
    if (phase) __syncthreads();
    stage_panel(sW, WmT + 256 + phase * 128, 512u);
    for (int ks = 0; ks < 4; ks++) {
      int k0 = phase * 128 + ks * 32;
      __builtin_prefetch(arow + 128 * 256 + k0, 0, 3);
      v16h a = ld_tile_f32(arow + k0 + kb);
#pragma unroll
      for (int n = 0; n < 16; n++) {
        v16h b = ld_tile_f16(sW + (size_t)(n * 16 + ln) * 128 + ks * 32 + kb);
        acc[n] = wmma_f16(a, b, acc[n]);
      }
    }
  }
#pragma unroll
  for (int r = 0; r < 8; r++) {
    int e = e0 + r + 8 * hi;
    int d = ei[E_ + e];
#pragma unroll
    for (int n = 0; n < 16; n++) {
      int col = n * 16 + ln;
      float m = acc[n][r] + Mj[(size_t)d * 256 + col];
      msg[(size_t)e * 256 + col] = (_Float16)m;
    }
  }
}

// ---------------------------------------------------------------------------
// Segment softmax: atomic max (monotone uint encoding), then exp + atomic sum.
// ---------------------------------------------------------------------------
__global__ __launch_bounds__(256) void k_segmax(const float* __restrict__ logits,
                                                const int* __restrict__ ei,
                                                unsigned* __restrict__ nmax) {
  int idx = blockIdx.x * 256 + threadIdx.x;
  if (idx >= E_ * 8) return;
  int e = idx >> 3, h = idx & 7;
  int s = ei[e];
  unsigned bits = __float_as_uint(logits[idx]);
  unsigned u = (bits & 0x80000000u) ? ~bits : (bits | 0x80000000u);
  atomicMax(&nmax[s * 8 + h], u);
}

__global__ __launch_bounds__(256) void k_segz(const float* __restrict__ logits,
                                              const int* __restrict__ ei,
                                              const unsigned* __restrict__ nmax,
                                              float* __restrict__ zbuf,
                                              float* __restrict__ nsum) {
  int idx = blockIdx.x * 256 + threadIdx.x;
  if (idx >= E_ * 8) return;
  int e = idx >> 3, h = idx & 7;
  int s = ei[e];
  unsigned u = nmax[s * 8 + h];
  unsigned fb = (u & 0x80000000u) ? (u ^ 0x80000000u) : ~u;
  float m = __uint_as_float(fb);
  float z = __expf(logits[idx] - m);
  zbuf[idx] = z;
  unsafeAtomicAdd(&nsum[s * 8 + h], z);
}

// agg[src] += msg * score  (one block per edge, one thread per col)
__global__ __launch_bounds__(256) void k_scatter(
    const _Float16* __restrict__ msg, const float* __restrict__ zbuf,
    const float* __restrict__ nsum, const int* __restrict__ ei,
    float* __restrict__ agg) {
  int e = blockIdx.x, c = threadIdx.x, h = c >> 5;
  int s = ei[e];
  float score = zbuf[(size_t)e * 8 + h] / nsum[s * 8 + h];
  float m = (float)msg[(size_t)e * 256 + c];
  unsafeAtomicAdd(&agg[(size_t)s * 256 + c], m * score);
}

// ---------------------------------------------------------------------------
// Continuous position bias: cpb[(b,q,k),h] = relu(rc@Wc1+bc1) @ Wc2 + bc2
// ---------------------------------------------------------------------------
__global__ __launch_bounds__(256) void k_cpb(
    const float* __restrict__ rc, const float* __restrict__ Wc1,
    const float* __restrict__ bc1, const float* __restrict__ Wc2,
    const float* __restrict__ bc2, float* __restrict__ cpb) {
  int idx = blockIdx.x * 256 + threadIdx.x;
  if (idx >= B_ * Q_ * K_) return;
  float x = rc[(size_t)idx * 2], y = rc[(size_t)idx * 2 + 1];
  float accv[8];
#pragma unroll
  for (int h = 0; h < 8; h++) accv[h] = bc2[h];
  for (int j = 0; j < CPB_; j++) {
    float hv = fmaf(y, Wc1[CPB_ + j], fmaf(x, Wc1[j], bc1[j]));
    hv = fmaxf(hv, 0.0f);
#pragma unroll
    for (int h = 0; h < 8; h++) accv[h] = fmaf(hv, Wc2[j * 8 + h], accv[h]);
  }
#pragma unroll
  for (int h = 0; h < 8; h++) cpb[(size_t)idx * 8 + h] = accv[h];
}

// V transpose for PV WMMA B-tiles: vT[b,h,hd,k] f16
__global__ __launch_bounds__(256) void k_vT(const float* __restrict__ kvbuf,
                                            _Float16* __restrict__ vT) {
  int idx = blockIdx.x * 256 + threadIdx.x;
  if (idx >= B_ * H_ * 32 * K_) return;
  int k = idx & (K_ - 1);
  int rest = idx >> 10;
  int hd = rest & 31;
  int bh = rest >> 5;
  int b = bh >> 3, h = bh & 7;
  vT[idx] = (_Float16)kvbuf[(size_t)(b * K_ + k) * 512 + 256 + h * 32 + hd];
}

// ---------------------------------------------------------------------------
// Flash cross-attention: one wave per (qtile16, h, b). QK^T and PV via WMMA,
// P staged through LDS (f16) to convert C-layout -> A-layout.
// Output stored in [B,H,Q,HD] order (faithful to reference's reshape bug).
// ---------------------------------------------------------------------------
__global__ __launch_bounds__(32) void k_attn(
    const float* __restrict__ qbuf, const float* __restrict__ kvbuf,
    const _Float16* __restrict__ vT, const float* __restrict__ cpb,
    const unsigned char* __restrict__ mask, float* __restrict__ obuf) {
  int qt = blockIdx.x, h = blockIdx.y, b = blockIdx.z;
  int lane = threadIdx.x & 31;
  int ln = lane & 15, hi = lane >> 4, kb = hi * 8;
  int q0 = qt * 16;
  __shared__ _Float16 pb[16 * 32];
  int qrow = q0 + ln; if (qrow > Q_ - 1) qrow = Q_ - 1;
  v16h aq = ld_tile_f32(qbuf + (size_t)(b * Q_ + qrow) * 256 + h * 32 + kb);
  float mrow[8], lrow[8];
  v8f o0 = {}, o1 = {};
#pragma unroll
  for (int r = 0; r < 8; r++) { mrow[r] = -3.0e38f; lrow[r] = 0.0f; }
  const float scale = 0.17677669529663687f;  // 1/sqrt(32)
  const _Float16* vbase = vT + (size_t)((b * H_ + h) * 32) * K_;
  for (int kc = 0; kc < K_; kc += 32) {
    v8f s0 = {}, s1 = {};
    v16h bk0 = ld_tile_f32(kvbuf + (size_t)(b * K_ + kc + ln) * 512 + h * 32 + kb);
    s0 = wmma_f16(aq, bk0, s0);
    v16h bk1 = ld_tile_f32(kvbuf + (size_t)(b * K_ + kc + 16 + ln) * 512 + h * 32 + kb);
    s1 = wmma_f16(aq, bk1, s1);
    int kcol0 = kc + ln, kcol1 = kc + 16 + ln;
    float msk0 = mask[b * K_ + kcol0] ? -1e30f : 0.0f;
    float msk1 = mask[b * K_ + kcol1] ? -1e30f : 0.0f;
#pragma unroll
    for (int r = 0; r < 8; r++) {
      int qq = q0 + r + 8 * hi; if (qq > Q_ - 1) qq = Q_ - 1;
      size_t cb = (size_t)(b * Q_ + qq) * K_ * 8;
      float v0 = s0[r] * scale + (msk0 < 0.f ? msk0 : cpb[cb + (size_t)kcol0 * 8 + h]);
      float v1 = s1[r] * scale + (msk1 < 0.f ? msk1 : cpb[cb + (size_t)kcol1 * 8 + h]);
      float mx = fmaxf(v0, v1);
      mx = fmaxf(mx, __shfl_xor(mx, 1, 32));
      mx = fmaxf(mx, __shfl_xor(mx, 2, 32));
      mx = fmaxf(mx, __shfl_xor(mx, 4, 32));
      mx = fmaxf(mx, __shfl_xor(mx, 8, 32));
      float mnew = fmaxf(mrow[r], mx);
      float corr = __expf(mrow[r] - mnew);
      float p0 = __expf(v0 - mnew);
      float p1 = __expf(v1 - mnew);
      float ps = p0 + p1;
      ps += __shfl_xor(ps, 1, 32);
      ps += __shfl_xor(ps, 2, 32);
      ps += __shfl_xor(ps, 4, 32);
      ps += __shfl_xor(ps, 8, 32);
      lrow[r] = lrow[r] * corr + ps;
      mrow[r] = mnew;
      o0[r] *= corr; o1[r] *= corr;
      pb[(r + 8 * hi) * 32 + ln]      = (_Float16)p0;
      pb[(r + 8 * hi) * 32 + 16 + ln] = (_Float16)p1;
    }
    asm volatile("s_wait_dscnt 0" ::: "memory");
    v16h ap  = ld_tile_f16(pb + ln * 32 + kb);
    v16h bv0 = ld_tile_f16(vbase + (size_t)ln * K_ + kc + kb);
    o0 = wmma_f16(ap, bv0, o0);
    v16h bv1 = ld_tile_f16(vbase + (size_t)(16 + ln) * K_ + kc + kb);
    o1 = wmma_f16(ap, bv1, o1);
  }
#pragma unroll
  for (int r = 0; r < 8; r++) {
    int qq = q0 + r + 8 * hi;
    if (qq < Q_) {
      float inv = 1.0f / lrow[r];
      size_t base = (size_t)((b * H_ + h) * Q_ + qq) * 32;
      obuf[base + ln]      = o0[r] * inv;
      obuf[base + 16 + ln] = o1[r] * inv;
    }
  }
}

// ---------------------------------------------------------------------------
// Host: orchestrate the full decoder layer on `stream`.
// ---------------------------------------------------------------------------
extern "C" void kernel_launch(void* const* d_in, const int* in_sizes, int n_in,
                              void* d_out, int out_size, void* d_ws, size_t ws_size,
                              hipStream_t stream) {
  const float* nodes  = (const float*)d_in[0];
  const float* images = (const float*)d_in[1];
  const float* rc     = (const float*)d_in[2];
  const float* ef     = (const float*)d_in[3];
  const int*   ei     = (const int*)d_in[4];
  const unsigned char* mask = (const unsigned char*)d_in[5];
  const float* ln1_g = (const float*)d_in[6];
  const float* ln1_b = (const float*)d_in[7];
  const float* ln2_g = (const float*)d_in[8];
  const float* ln2_b = (const float*)d_in[9];
  const float* ln3_g = (const float*)d_in[10];
  const float* ln3_b = (const float*)d_in[11];
  const float* Wni = (const float*)d_in[12];
  const float* bni = (const float*)d_in[13];
  const float* Wnj = (const float*)d_in[14];
  const float* bnj = (const float*)d_in[15];
  const float* We  = (const float*)d_in[16];
  const float* be  = (const float*)d_in[17];
  const float* attn_proj = (const float*)d_in[18];
  const float* Wmsg = (const float*)d_in[19];
  const float* bmsg = (const float*)d_in[20];
  const float* Wg_out = (const float*)d_in[21];
  const float* bg_out = (const float*)d_in[22];
  const float* Wq  = (const float*)d_in[23];
  const float* bq  = (const float*)d_in[24];
  const float* Wkv = (const float*)d_in[25];
  const float* bkv = (const float*)d_in[26];
  const float* Wc1 = (const float*)d_in[27];
  const float* bc1 = (const float*)d_in[28];
  const float* Wc2 = (const float*)d_in[29];
  const float* bc2 = (const float*)d_in[30];
  const float* Wc_out = (const float*)d_in[31];
  const float* bc_out = (const float*)d_in[32];
  const float* Wf1 = (const float*)d_in[33];
  const float* bf1 = (const float*)d_in[34];
  const float* Wf2 = (const float*)d_in[35];
  const float* bf2 = (const float*)d_in[36];

  // Workspace bump allocator (256B aligned)
  char* wsb = (char*)d_ws;
  size_t off = 0;
  auto alloc = [&](size_t bytes) -> void* {
    void* p = wsb + off;
    off += (bytes + 255) & ~(size_t)255;
    return p;
  };
  _Float16* WniT = (_Float16*)alloc(256 * 256 * 2);
  _Float16* WnjT = (_Float16*)alloc(256 * 256 * 2);
  _Float16* WeT  = (_Float16*)alloc(256 * 256 * 2);
  _Float16* WmT  = (_Float16*)alloc(256 * 512 * 2);   // Wmsg^T : 256 x 512
  _Float16* WgoT = (_Float16*)alloc(256 * 256 * 2);
  _Float16* WqT  = (_Float16*)alloc(256 * 256 * 2);
  _Float16* WkvT = (_Float16*)alloc(512 * 256 * 2);   // 512 x 256
  _Float16* WcoT = (_Float16*)alloc(256 * 256 * 2);
  _Float16* Wf1T = (_Float16*)alloc(1024 * 256 * 2);  // 1024 x 256
  _Float16* Wf2T = (_Float16*)alloc(256 * 1024 * 2);  // 256 x 1024
  float* biasH  = (float*)alloc(256 * 4);
  float* xbuf   = (float*)alloc((size_t)N_ * 256 * 4);
  float* Pi     = (float*)alloc((size_t)N_ * 256 * 4);
  float* Pj     = (float*)alloc((size_t)N_ * 256 * 4);
  float* Mj     = (float*)alloc((size_t)N_ * 256 * 4);
  float* logits = (float*)alloc((size_t)E_ * 8 * 4);
  unsigned* nmax = (unsigned*)alloc((size_t)N_ * 8 * 4);
  float* nsum   = (float*)alloc((size_t)N_ * 8 * 4);
  float* zbuf   = (float*)alloc((size_t)E_ * 8 * 4);
  _Float16* msg = (_Float16*)alloc((size_t)E_ * 256 * 2);
  float* agg    = (float*)alloc((size_t)N_ * 256 * 4);
  float* nodes1 = (float*)alloc((size_t)N_ * 256 * 4);
  float* x2     = (float*)alloc((size_t)N_ * 256 * 4);
  float* qbuf   = (float*)alloc((size_t)N_ * 256 * 4);
  float* kvbuf  = (float*)alloc((size_t)B_ * K_ * 512 * 4);
  _Float16* vT  = (_Float16*)alloc((size_t)B_ * H_ * 32 * K_ * 2);
  float* cpbbuf = (float*)alloc((size_t)B_ * Q_ * K_ * 8 * 4);
  float* obuf   = (float*)alloc((size_t)N_ * 256 * 4);
  float* nodes2 = (float*)alloc((size_t)N_ * 256 * 4);
  float* x3     = (float*)alloc((size_t)N_ * 256 * 4);
  float* ffbuf  = (float*)alloc((size_t)N_ * 1024 * 4);

  auto tc = [&](const float* src, _Float16* dst, int K, int N) {
    k_transpose_cvt<<<(K * N + 255) / 256, 256, 0, stream>>>(src, dst, K, N);
  };
  auto gemm = [&](const float* A, int lda, const _Float16* Bt, int ldb,
                  const float* bias, const float* resid, int ldr,
                  float* C, int ldc, int M, int Ncols, int ksteps, int act) {
    int panels = Ncols / 256;
    int waves = ((M + 15) / 16) * panels;
    int blocks = (waves + 7) / 8;
    k_gemm<<<blocks, 256, 0, stream>>>(A, lda, Bt, ldb, bias, resid, ldr,
                                       C, ldc, M, panels, ksteps, act);
  };

  // ---- weight prep ----
  tc(Wni, WniT, 256, 256);
  tc(Wnj, WnjT, 256, 256);
  tc(We,  WeT,  256, 256);
  tc(Wmsg, WmT, 512, 256);
  tc(Wg_out, WgoT, 256, 256);
  tc(Wq,  WqT,  256, 256);
  tc(Wkv, WkvT, 256, 512);
  tc(Wc_out, WcoT, 256, 256);
  tc(Wf1, Wf1T, 256, 1024);
  tc(Wf2, Wf2T, 1024, 256);
  k_add3<<<1, 256, 0, stream>>>(bni, bnj, be, biasH);

  // ---- graph attention ----
  k_ln<<<N_, 256, 0, stream>>>(nodes, ln1_g, ln1_b, xbuf);
  gemm(xbuf, 256, WniT, 256, biasH, nullptr, 0, Pi, 256, N_, 256, 8, 0);
  gemm(xbuf, 256, WnjT, 256, nullptr, nullptr, 0, Pj, 256, N_, 256, 8, 0);
  gemm(xbuf, 256, WmT,  512, bmsg,   nullptr, 0, Mj, 256, N_, 256, 8, 0);
  k_edge_hidden<<<E_ / 128, 256, 0, stream>>>(ef, WeT, Pi, Pj, ei, attn_proj, logits);
  (void)hipMemsetAsync(nmax, 0, (size_t)N_ * 8 * 4, stream);
  (void)hipMemsetAsync(nsum, 0, (size_t)N_ * 8 * 4, stream);
  (void)hipMemsetAsync(agg,  0, (size_t)N_ * 256 * 4, stream);
  k_segmax<<<(E_ * 8) / 256, 256, 0, stream>>>(logits, ei, nmax);
  k_segz<<<(E_ * 8) / 256, 256, 0, stream>>>(logits, ei, nmax, zbuf, nsum);
  k_edge_msg<<<E_ / 128, 256, 0, stream>>>(ef, WmT, Mj, ei, msg);
  k_scatter<<<E_, 256, 0, stream>>>(msg, zbuf, nsum, ei, agg);
  gemm(agg, 256, WgoT, 256, bg_out, nodes, 256, nodes1, 256, N_, 256, 8, 0);

  // ---- cross attention ----
  k_ln<<<N_, 256, 0, stream>>>(nodes1, ln2_g, ln2_b, x2);
  gemm(x2, 256, WqT, 256, bq, nullptr, 0, qbuf, 256, N_, 256, 8, 0);
  gemm(images, 256, WkvT, 256, bkv, nullptr, 0, kvbuf, 512, B_ * K_, 512, 8, 0);
  k_vT<<<(B_ * H_ * 32 * K_ + 255) / 256, 256, 0, stream>>>(kvbuf, vT);
  k_cpb<<<(B_ * Q_ * K_ + 255) / 256, 256, 0, stream>>>(rc, Wc1, bc1, Wc2, bc2, cpbbuf);
  k_attn<<<dim3((Q_ + 15) / 16, H_, B_), 32, 0, stream>>>(qbuf, kvbuf, vT, cpbbuf,
                                                          mask, obuf);
  gemm(obuf, 256, WcoT, 256, bc_out, nodes1, 256, nodes2, 256, N_, 256, 8, 0);

  // ---- FFN (residual from normed input, as in reference) ----
  k_ln<<<N_, 256, 0, stream>>>(nodes2, ln3_g, ln3_b, x3);
  gemm(x3, 256, Wf1T, 256, bf1, nullptr, 0, ffbuf, 1024, N_, 1024, 8, 1);
  gemm(ffbuf, 1024, Wf2T, 1024, bf2, x3, 256, (float*)d_out, 256, N_, 256, 32, 0);
}
